// LearnedFeatureExtractor_90993177133468
// MI455X (gfx1250) — matrix operand order repaired and checked by reading, not compiled, not run on hardware
//
#include <hip/hip_runtime.h>

// LearnedFeatureExtractor for MI455X (gfx1250, wave32).
// Block = 1 molecule (L=64 atoms, D=128), 256 threads = 8 waves, 128KB LDS arena.
// All GEMMs via v_wmma_f32_16x16x32_bf16 (f32 accumulate).
// Schedule: each wave owns one 16-wide output-column tile (nt == wave) so the
// weight B-operands are loop-invariant (hoisted) and all stores are branch-free.

#define LATOMS 64
#define DDIM   128

typedef __attribute__((ext_vector_type(16))) __bf16 bf16x16;
typedef __attribute__((ext_vector_type(8)))  __bf16 bf16x8;
typedef __attribute__((ext_vector_type(8)))  float  f32x8;

// A-operand (16x32 tile, M = lane&15): lanes 0-15 hold K = k0+{0..7, 16..23},
// lanes 16-31 hold K = k0+{8..15, 24..31}. Two 16B loads + shuffle.
__device__ __forceinline__ bf16x16 load_a(const __bf16* row, int k0, int kb8) {
    bf16x8 lo = *(const bf16x8*)(row + k0 + kb8);
    bf16x8 hi = *(const bf16x8*)(row + k0 + 16 + kb8);
    return __builtin_shufflevector(lo, hi, 0,1,2,3,4,5,6,7,8,9,10,11,12,13,14,15);
}

// B-operand (32x16 tile, N = lane&15): lanes 0-15 hold K = k0+{0..7, 16..23},
// lanes 16-31 hold K = k0+{8..15, 24..31}. Two 16B loads + shuffle.
__device__ __forceinline__ bf16x16 load_b(const __bf16* row, int k0, int half) {
    bf16x8 lo = *(const bf16x8*)(row + k0 + half * 8);
    bf16x8 hi = *(const bf16x8*)(row + k0 + 16 + half * 8);
    return __builtin_shufflevector(lo, hi, 0,1,2,3,4,5,6,7,8,9,10,11,12,13,14,15);
}

__device__ __forceinline__ f32x8 wmma_bf16(bf16x16 a, bf16x16 b, f32x8 c) {
    return __builtin_amdgcn_wmma_f32_16x16x32_bf16(
        /*neg_a=*/false, a, /*neg_b=*/false, b,
        /*c_mod=*/(short)0, c, /*reuse_a=*/false, /*reuse_b=*/false);
}

__global__ void prep_weights(const float* __restrict__ Wi,   // [384,128]
                             const float* __restrict__ Wo,   // [128,128]
                             __bf16* __restrict__ wb) {
    int i = blockIdx.x * 256 + threadIdx.x;                  // grid covers 65536
    if (i < 49152) wb[i] = (__bf16)Wi[i];
    else           wb[i] = (__bf16)Wo[i - 49152];
}

__global__ void __launch_bounds__(256)
lfe_kernel(const float* __restrict__ xg,      // [B*L, D] f32
           const float* __restrict__ b_in,    // [384]
           const float* __restrict__ b_out,   // [128]
           const float* __restrict__ Wpool,   // [4,16,128]
           const float* __restrict__ bpool,   // [4,16]
           const __bf16* __restrict__ wbin,   // [384,128] bf16
           const __bf16* __restrict__ wbout,  // [128,128] bf16
           float* __restrict__ out)           // [B, 64]
{
    __shared__ __align__(16) unsigned char smem[131072];
    float*  xf    = (float*) (smem + 0);        // [64][128] f32   (live through E)
    __bf16* xb    = (__bf16*)(smem + 32768);    // [64][128] bf16  (dead after B)
    __bf16* qb    = (__bf16*)(smem + 49152);    // [64][128] bf16, pre-scaled 1/sqrt(32)
    __bf16* kbf   = (__bf16*)(smem + 65536);    // [64][128] bf16
    __bf16* vt    = (__bf16*)(smem + 81920);    // [128][64] bf16 (V transposed)
    __bf16* ctxb  = (__bf16*)(smem + 98304);    // [64][128] bf16
    __bf16* pscr  = (__bf16*)(smem + 114688);   // 8 waves x [16][64] bf16
    float*  att   = (float*) (smem + 32768);    // [64][128] f32, overlays xb+qb (phase D+)
    float*  dotv  = (float*) (smem + 114688);   // [64] f32, overlays pscr (phase E)
    float*  aw    = (float*) (smem + 114944);   // [64] f32
    float*  pooled= (float*) (smem + 115200);   // [4][128] f32

    const int t    = threadIdx.x;
    const int lane = t & 31;
    const int wave = t >> 5;
    const int half = lane >> 4;
    const int l16  = lane & 15;
    const int kb8  = half * 8;
    const int bmol = blockIdx.x;

    const float* xsrc = xg + (size_t)bmol * (LATOMS * DDIM);

    // Warm L2/L0 with the (shared) bf16 weights.
    if (t < 128) __builtin_prefetch(wbin + t * 384, 0, 0);

    // ---- Phase A: stage x into LDS as f32 + bf16 ----
    #pragma unroll
    for (int i = 0; i < 8; ++i) {
        int idx = (i * 256 + t) * 4;
        float4 v = *(const float4*)(xsrc + idx);
        *(float4*)(xf + idx) = v;
        xb[idx + 0] = (__bf16)v.x; xb[idx + 1] = (__bf16)v.y;
        xb[idx + 2] = (__bf16)v.z; xb[idx + 3] = (__bf16)v.w;
    }
    __syncthreads();

    // ---- Phase B: QKV = x @ W_in^T + b_in ----
    // Wave owns weight-column tile (wave*16); B operands hoisted out of the mt loop.
    {
        // Q tiles (e in [0,128)), scaled by 1/sqrt(dh) before bf16 round.
        const int e = wave * 16 + l16;
        const float bias = b_in[e];
        const __bf16* brow = wbin + (size_t)e * DDIM;
        bf16x16 w0 = load_b(brow,  0, half), w1 = load_b(brow, 32, half);
        bf16x16 w2 = load_b(brow, 64, half), w3 = load_b(brow, 96, half);
        for (int mt = 0; mt < 4; ++mt) {
            f32x8 c;
            #pragma unroll
            for (int r = 0; r < 8; ++r) c[r] = bias;
            const __bf16* arow = xb + (mt * 16 + l16) * DDIM;
            c = wmma_bf16(load_a(arow,  0, kb8), w0, c);
            c = wmma_bf16(load_a(arow, 32, kb8), w1, c);
            c = wmma_bf16(load_a(arow, 64, kb8), w2, c);
            c = wmma_bf16(load_a(arow, 96, kb8), w3, c);
            #pragma unroll
            for (int r = 0; r < 8; ++r)
                qb[(mt * 16 + r + half * 8) * DDIM + e] = (__bf16)(c[r] * 0.1767766953f);
        }
    }
    {
        // K tiles (e in [128,256))
        const int ec = wave * 16 + l16;
        const float bias = b_in[128 + ec];
        const __bf16* brow = wbin + (size_t)(128 + ec) * DDIM;
        bf16x16 w0 = load_b(brow,  0, half), w1 = load_b(brow, 32, half);
        bf16x16 w2 = load_b(brow, 64, half), w3 = load_b(brow, 96, half);
        for (int mt = 0; mt < 4; ++mt) {
            f32x8 c;
            #pragma unroll
            for (int r = 0; r < 8; ++r) c[r] = bias;
            const __bf16* arow = xb + (mt * 16 + l16) * DDIM;
            c = wmma_bf16(load_a(arow,  0, kb8), w0, c);
            c = wmma_bf16(load_a(arow, 32, kb8), w1, c);
            c = wmma_bf16(load_a(arow, 64, kb8), w2, c);
            c = wmma_bf16(load_a(arow, 96, kb8), w3, c);
            #pragma unroll
            for (int r = 0; r < 8; ++r)
                kbf[(mt * 16 + r + half * 8) * DDIM + ec] = (__bf16)c[r];
        }
    }
    {
        // V tiles (e in [256,384)), stored transposed [d][l].
        const int ec = wave * 16 + l16;
        const float bias = b_in[256 + ec];
        const __bf16* brow = wbin + (size_t)(256 + ec) * DDIM;
        bf16x16 w0 = load_b(brow,  0, half), w1 = load_b(brow, 32, half);
        bf16x16 w2 = load_b(brow, 64, half), w3 = load_b(brow, 96, half);
        for (int mt = 0; mt < 4; ++mt) {
            f32x8 c;
            #pragma unroll
            for (int r = 0; r < 8; ++r) c[r] = bias;
            const __bf16* arow = xb + (mt * 16 + l16) * DDIM;
            c = wmma_bf16(load_a(arow,  0, kb8), w0, c);
            c = wmma_bf16(load_a(arow, 32, kb8), w1, c);
            c = wmma_bf16(load_a(arow, 64, kb8), w2, c);
            c = wmma_bf16(load_a(arow, 96, kb8), w3, c);
            #pragma unroll
            for (int r = 0; r < 8; ++r)
                vt[ec * LATOMS + (mt * 16 + r + half * 8)] = (__bf16)c[r];
        }
    }
    __syncthreads();

    // ---- Phase C: per (head, m-tile): S = Q K^T, softmax, ctx = P V ----
    __bf16* pw = pscr + wave * (16 * 64);
    for (int it = wave; it < 16; it += 8) {
        const int h = it >> 2, mt = it & 3;
        f32x8 s[4];
        #pragma unroll
        for (int j = 0; j < 4; ++j)
            #pragma unroll
            for (int r = 0; r < 8; ++r) s[j][r] = 0.f;
        const __bf16* arow = qb + (mt * 16 + l16) * DDIM + h * 32;
        bf16x16 aq = load_a(arow, 0, kb8);
        #pragma unroll
        for (int j = 0; j < 4; ++j) {
            const __bf16* brow = kbf + (j * 16 + l16) * DDIM + h * 32;
            s[j] = wmma_bf16(aq, load_b(brow, 0, half), s[j]);
        }
        // Row softmax: row (mt*16 + r + 8*half), 64 cols = 4 accums x 16 lanes.
        #pragma unroll
        for (int r = 0; r < 8; ++r) {
            float m = fmaxf(fmaxf(s[0][r], s[1][r]), fmaxf(s[2][r], s[3][r]));
            m = fmaxf(m, __shfl_xor(m, 1, 32));
            m = fmaxf(m, __shfl_xor(m, 2, 32));
            m = fmaxf(m, __shfl_xor(m, 4, 32));
            m = fmaxf(m, __shfl_xor(m, 8, 32));
            float p0 = __expf(s[0][r] - m), p1 = __expf(s[1][r] - m);
            float p2 = __expf(s[2][r] - m), p3 = __expf(s[3][r] - m);
            float sum = p0 + p1 + p2 + p3;
            sum += __shfl_xor(sum, 1, 32);
            sum += __shfl_xor(sum, 2, 32);
            sum += __shfl_xor(sum, 4, 32);
            sum += __shfl_xor(sum, 8, 32);
            const float inv = 1.0f / sum;
            const int row = r + half * 8;
            pw[row * 64 +  0 + l16] = (__bf16)(p0 * inv);
            pw[row * 64 + 16 + l16] = (__bf16)(p1 * inv);
            pw[row * 64 + 32 + l16] = (__bf16)(p2 * inv);
            pw[row * 64 + 48 + l16] = (__bf16)(p3 * inv);
        }
        // Cross-lane LDS RAW within the wave: wait on DS counter.
        asm volatile("s_wait_dscnt 0" ::: "memory");
        // ctx[16, 32] = P[16,64] @ V[64,32-slice]
        #pragma unroll
        for (int t2 = 0; t2 < 2; ++t2) {
            f32x8 c2;
            #pragma unroll
            for (int r = 0; r < 8; ++r) c2[r] = 0.f;
            const int dcol = h * 32 + t2 * 16 + l16;
            #pragma unroll
            for (int ks = 0; ks < 2; ++ks)
                c2 = wmma_bf16(load_a(pw + l16 * 64, ks * 32, kb8),
                               load_b(vt + dcol * LATOMS, ks * 32, half), c2);
            #pragma unroll
            for (int r = 0; r < 8; ++r)
                ctxb[(mt * 16 + r + half * 8) * DDIM + dcol] = (__bf16)c2[r];
        }
    }
    __syncthreads();

    // ---- Phase D: attended = ctx @ W_out^T + b_out ----
    // Wave owns column tile (wave*16); W_out B operands hoisted.
    {
        const int e = wave * 16 + l16;
        const float bias = b_out[e];
        const __bf16* brow = wbout + (size_t)e * DDIM;
        bf16x16 w0 = load_b(brow,  0, half), w1 = load_b(brow, 32, half);
        bf16x16 w2 = load_b(brow, 64, half), w3 = load_b(brow, 96, half);
        for (int mt = 0; mt < 4; ++mt) {
            f32x8 c;
            #pragma unroll
            for (int r = 0; r < 8; ++r) c[r] = bias;
            const __bf16* arow = ctxb + (mt * 16 + l16) * DDIM;
            c = wmma_bf16(load_a(arow,  0, kb8), w0, c);
            c = wmma_bf16(load_a(arow, 32, kb8), w1, c);
            c = wmma_bf16(load_a(arow, 64, kb8), w2, c);
            c = wmma_bf16(load_a(arow, 96, kb8), w3, c);
            #pragma unroll
            for (int r = 0; r < 8; ++r)
                att[(mt * 16 + r + half * 8) * DDIM + e] = c[r];
        }
    }
    __syncthreads();

    // ---- Phase E1: dot(attended[l], x[l]) with 4 threads per row ----
    {
        const int l = t >> 2, p = t & 3;
        const float* ar = att + l * DDIM + p * 32;
        const float* xr = xf  + l * DDIM + p * 32;
        float s = 0.f;
        #pragma unroll
        for (int j = 0; j < 32; ++j) s += ar[j] * xr[j];
        s += __shfl_xor(s, 1, 32);
        s += __shfl_xor(s, 2, 32);
        if (p == 0) dotv[l] = s;
    }
    __syncthreads();
    // ---- E2: softmax over the 64 row-scores ----
    if (t < 64) {
        float m = -3.4e38f;
        for (int j = 0; j < 64; ++j) m = fmaxf(m, dotv[j]);
        float sum = 0.f;
        for (int j = 0; j < 64; ++j) sum += __expf(dotv[j] - m);
        aw[t] = __expf(dotv[t] - m) / sum;
    }
    __syncthreads();
    // ---- E3: 4 pools per feature column ----
    if (t < 128) {
        float wm = 0.f, mx = -3.4e38f, sm = 0.f, s2 = 0.f;
        for (int l = 0; l < 64; ++l) {
            float a = att[l * DDIM + t];
            wm += a * aw[l];
            mx = fmaxf(mx, a);
            sm += a;
            s2 += a * a;
        }
        float mean = sm * (1.f / 64.f);
        float var  = (s2 - sm * sm * (1.f / 64.f)) * (1.f / 63.f);   // ddof=1
        pooled[0 * DDIM + t] = wm;
        pooled[1 * DDIM + t] = mx;
        pooled[2 * DDIM + t] = mean;
        pooled[3 * DDIM + t] = sqrtf(fmaxf(var, 0.f));
    }
    __syncthreads();
    // ---- E4: 4 tiny projections (16 outputs each) ----
    if (t < 64) {
        const int p = t >> 4, f = t & 15;
        const float* wp = Wpool + (p * 16 + f) * DDIM;
        float acc = bpool[p * 16 + f];
        #pragma unroll 4
        for (int d = 0; d < DDIM; ++d) acc += pooled[p * DDIM + d] * wp[d];
        out[(size_t)bmol * 64 + t] = acc;
    }
}

extern "C" void kernel_launch(void* const* d_in, const int* in_sizes, int n_in,
                              void* d_out, int out_size, void* d_ws, size_t ws_size,
                              hipStream_t stream) {
    (void)in_sizes; (void)n_in; (void)out_size; (void)ws_size;
    const float* x      = (const float*)d_in[0];
    // d_in[1] = batch_indices (int64): unused, segments are equal-length.
    const float* W_in   = (const float*)d_in[2];
    const float* b_in   = (const float*)d_in[3];
    const float* W_out  = (const float*)d_in[4];
    const float* b_out  = (const float*)d_in[5];
    const float* W_pool = (const float*)d_in[6];
    const float* b_pool = (const float*)d_in[7];

    __bf16* wb = (__bf16*)d_ws;           // [0,49152): W_in bf16; [49152,65536): W_out bf16
    prep_weights<<<256, 256, 0, stream>>>(W_in, W_out, wb);
    lfe_kernel<<<8192, 256, 0, stream>>>(x, b_in, b_out, W_pool, b_pool,
                                         wb, wb + 49152, (float*)d_out);
}
